// GPT_65962107732225
// MI455X (gfx1250) — matrix operand order repaired and checked by reading, not compiled, not run on hardware
//
#include <hip/hip_runtime.h>
#include <hip/hip_bf16.h>
#include <cstdint>
#include <cstddef>

// ---------------------------------------------------------------------------
// GPT forward for MI455X (gfx1250).  All contractions run on
// v_wmma_f32_16x16x32_bf16 (f32 accumulate).  Residual stream fp32,
// GEMM operands bf16 (weights converted once per call into d_ws).
// GEMM/attention tile staging uses the CDNA5 async memory->LDS path
// (GLOBAL_LOAD_ASYNC_TO_LDS_B128, ASYNCcnt) with double-buffered LDS.
// ---------------------------------------------------------------------------

typedef __attribute__((ext_vector_type(4)))  __bf16 bf16x4;
typedef __attribute__((ext_vector_type(8)))  __bf16 bf16x8;
typedef __attribute__((ext_vector_type(16))) __bf16 bf16x16;
typedef __attribute__((ext_vector_type(8)))  float  floatx8;

#define DEV_INLINE __device__ __forceinline__

constexpr int B_ = 2, T_ = 1024, V_ = 32000, D_ = 1024, H_ = 16, L_ = 8;
constexpr int DH_ = 64, FF_ = 4096, BT_ = B_ * T_;

DEV_INLINE bf16x16 cat8(bf16x8 lo, bf16x8 hi) {
  return __builtin_shufflevector(lo, hi, 0, 1, 2, 3, 4, 5, 6, 7,
                                 8, 9, 10, 11, 12, 13, 14, 15);
}
DEV_INLINE floatx8 wmma_bf16(bf16x16 a, bf16x16 b, floatx8 c) {
  // (neg_a, A, neg_b, B, c_mod, C, reuse_a, reuse_b)
  return __builtin_amdgcn_wmma_f32_16x16x32_bf16(false, a, false, b,
                                                 (short)0, c, false, false);
}

// CDNA5 async memory->LDS copy, 16B per lane.  VDST supplies the per-lane
// LDS byte address, VADDR the 64-bit global address (GV mode).  Tracked by
// ASYNCcnt.  (Flat-pointer low 32 bits == LDS byte offset per the aperture
// mapping, so truncating a generic pointer to 32 bits is the dsaddr.)
DEV_INLINE void async_load_b128(uint32_t lds_off, const void* gaddr) {
  asm volatile("global_load_async_to_lds_b128 %0, %1, off"
               :: "v"(lds_off), "v"(gaddr)
               : "memory");
}
DEV_INLINE void wait_async0() {
  asm volatile("s_wait_asynccnt 0x0" ::: "memory");
}
DEV_INLINE uint32_t lds_addr(const void* p) {
  return (uint32_t)(uintptr_t)p;
}

// ---------------------------------------------------------------------------
// fp32 -> bf16 bulk convert (weights), float4-vectorized grid-stride
// ---------------------------------------------------------------------------
__global__ __launch_bounds__(256) void k_cast_bf16(const float* __restrict__ in,
                                                   __bf16* __restrict__ out,
                                                   size_t n4) {
  size_t i = (size_t)blockIdx.x * blockDim.x + threadIdx.x;
  const size_t stride = (size_t)gridDim.x * blockDim.x;
  for (; i < n4; i += stride) {
    float4 v = ((const float4*)in)[i];
    bf16x4 o = {(__bf16)v.x, (__bf16)v.y, (__bf16)v.z, (__bf16)v.w};
    ((bf16x4*)out)[i] = o;
  }
}

// ---------------------------------------------------------------------------
// embedding gather: x[bt,:] = embedding[idx[bt],:]   (one block per row)
// ---------------------------------------------------------------------------
__global__ __launch_bounds__(256) void k_embed(const int* __restrict__ idx,
                                               const float* __restrict__ emb,
                                               float* __restrict__ x) {
  const int row = blockIdx.x;
  const int tid = threadIdx.x;  // 256 threads * float4 == 1024 floats
  const int e = idx[row];
  ((float4*)(x + (size_t)row * D_))[tid] =
      ((const float4*)(emb + (size_t)e * D_))[tid];
}

// ---------------------------------------------------------------------------
// RMSNorm (D=1024) -> bf16 operand for the following GEMM
// ---------------------------------------------------------------------------
__global__ __launch_bounds__(256) void k_rmsnorm(const float* __restrict__ x,
                                                 __bf16* __restrict__ outb) {
  __shared__ float red[256];
  const int row = blockIdx.x, tid = threadIdx.x;
  float4 v = ((const float4*)(x + (size_t)row * D_))[tid];
  red[tid] = v.x * v.x + v.y * v.y + v.z * v.z + v.w * v.w;
  __syncthreads();
  for (int s = 128; s > 0; s >>= 1) {
    if (tid < s) red[tid] += red[tid + s];
    __syncthreads();
  }
  const float scale = rsqrtf(red[0] * (1.0f / D_) + 1e-6f);
  bf16x4 o = {(__bf16)(v.x * scale), (__bf16)(v.y * scale),
              (__bf16)(v.z * scale), (__bf16)(v.w * scale)};
  ((bf16x4*)(outb + (size_t)row * D_))[tid] = o;
}

// ---------------------------------------------------------------------------
// WMMA GEMM:  C[M,N] = A[M,K] * W[N,K]^T      (both operands K-contiguous bf16)
//   block tile 128x64, BK=64, 8 waves in 4(M) x 2(N), each wave 32x32.
//   LDS rows padded to 72 elements: 144B stride = 36 banks, gcd(36,64)=4 ->
//   16 lanes hit 16 distinct banks; stride is a 16B multiple (ds_load_b128).
//   Double-buffered staging via GLOBAL_LOAD_ASYNC_TO_LDS_B128: issue tile
//   it+1 async, run 16 ds_load_b128 + 8 WMMA on tile it, s_wait_asynccnt 0,
//   one barrier.
// Epilogue:  EPI=0 fp32 store | EPI=1 residual += | EPI=2 relu^2 -> bf16
// ---------------------------------------------------------------------------
constexpr int GBM = 128, GBN = 64, GBK = 64, LP = 72;

template <int EPI>
__global__ __launch_bounds__(256) void k_gemm_bf16(
    const __bf16* __restrict__ A, const __bf16* __restrict__ W, int M, int N,
    int K, float* __restrict__ resid, float* __restrict__ outF,
    __bf16* __restrict__ outB) {
  __shared__ __bf16 sA[2][GBM * LP];  // 2 x 18 KB
  __shared__ __bf16 sB[2][GBN * LP];  // 2 x 9 KB

  const int tid = threadIdx.x;
  const int lane = tid & 31;
  const int wave = tid >> 5;
  const int wm = wave & 3;   // 4 waves along M
  const int wn = wave >> 2;  // 2 waves along N
  const int lr = lane & 15;
  const int hs = lane >> 4;  // half-wave select
  const int blockN = blockIdx.x * GBN;
  const int blockM = blockIdx.y * GBM;
  const int kloA = hs * 8;   // A-frag K base (lanes 16-31 hold K 8..15/24..31)
  const int khB = hs * 8;    // B-frag K base (same as A: K 8..15/24..31 for hs=1)

  // per-thread staging coordinates (fixed): A 128x64 -> 32 elems/thread,
  // W 64x64 -> 16 elems/thread
  const int arow = tid >> 1, ach = (tid & 1) * 32;
  const int brow = tid >> 2, bch = (tid & 3) * 16;

  auto stage = [&](int buf, int k0) {
    const __bf16* ga = A + (size_t)(blockM + arow) * K + k0 + ach;
    const uint32_t la = lds_addr(&sA[buf][arow * LP + ach]);
#pragma unroll
    for (int e = 0; e < 4; ++e) async_load_b128(la + e * 16, ga + e * 8);
    const __bf16* gb = W + (size_t)(blockN + brow) * K + k0 + bch;
    const uint32_t lb = lds_addr(&sB[buf][brow * LP + bch]);
#pragma unroll
    for (int e = 0; e < 2; ++e) async_load_b128(lb + e * 16, gb + e * 8);
  };

  floatx8 acc[2][2] = {};

  stage(0, 0);
  wait_async0();
  __syncthreads();

  const int nk = K / GBK;
  for (int it = 0; it < nk; ++it) {
    const int cur = it & 1;
    if (it + 1 < nk) stage(1 - cur, (it + 1) * GBK);  // overlap with compute

#pragma unroll
    for (int ks = 0; ks < 2; ++ks) {  // two 16x16x32 K-steps per tile
      bf16x16 af[2], bfg[2];
#pragma unroll
      for (int i = 0; i < 2; ++i) {
        const __bf16* p =
            &sA[cur][(wm * 32 + i * 16 + lr) * LP + ks * 32 + kloA];
        af[i] = cat8(*(const bf16x8*)p, *(const bf16x8*)(p + 16));
      }
#pragma unroll
      for (int j = 0; j < 2; ++j) {
        const __bf16* p =
            &sB[cur][(wn * 32 + j * 16 + lr) * LP + ks * 32 + khB];
        bfg[j] = cat8(*(const bf16x8*)p, *(const bf16x8*)(p + 16));
      }
#pragma unroll
      for (int i = 0; i < 2; ++i)
#pragma unroll
        for (int j = 0; j < 2; ++j)
          acc[i][j] = wmma_bf16(af[i], bfg[j], acc[i][j]);
    }

    if (it + 1 < nk) wait_async0();  // next tile fully in LDS
    __syncthreads();
  }

  // C layout: lane -> col = lr (+16*j), row = hs*8 + v (+16*i)
  const int colBase = blockN + wn * 32;
  const int rowBase = blockM + wm * 32 + hs * 8;
#pragma unroll
  for (int i = 0; i < 2; ++i)
#pragma unroll
    for (int j = 0; j < 2; ++j) {
      const int col = colBase + j * 16 + lr;
#pragma unroll
      for (int v = 0; v < 8; ++v) {
        const size_t idx = (size_t)(rowBase + i * 16 + v) * N + col;
        const float val = acc[i][j][v];
        if (EPI == 0) {
          outF[idx] = val;
        } else if (EPI == 1) {
          resid[idx] += val;
        } else {
          const float r = fmaxf(val, 0.0f);
          outB[idx] = (__bf16)(r * r);
        }
      }
    }
}

// ---------------------------------------------------------------------------
// RoPE split: qkv fp32 [BT][3,H,DH] -> Q,K (roped, Q pre-scaled by 1/sqrt(DH))
// and V, all bf16 in [B,H,T,DH].  Tables computed inline: quarter=16 real
// frequencies, quarter zero-padded (cos=1, sin=0) as in the reference.
// ---------------------------------------------------------------------------
__global__ __launch_bounds__(64) void k_rope_split(const float* __restrict__ qkv,
                                                   __bf16* __restrict__ Q,
                                                   __bf16* __restrict__ Kk,
                                                   __bf16* __restrict__ Vv) {
  const int blk = blockIdx.x;  // (b*H + h)*T + t
  const int t = blk % T_;
  const int bh = blk / T_;
  const int h = bh % H_;
  const int tid = threadIdx.x;  // 64
  const size_t inrow = (size_t)((bh / H_) * T_ + t) * (3 * D_);
  const size_t orow = (size_t)bh * T_ * DH_ + (size_t)t * DH_;

  // V: straight bf16 copy (64 lanes, one element each)
  Vv[orow + tid] = (__bf16)qkv[inrow + 2 * D_ + h * DH_ + tid];

  const int grp = tid >> 5;  // 0 = q, 1 = k
  const int i = tid & 31;    // rotation pair index
  const float x1 = qkv[inrow + grp * D_ + h * DH_ + i];
  const float x2 = qkv[inrow + grp * D_ + h * DH_ + 32 + i];
  float c, s;
  if (i < 16) {
    const float f = __expf(-(float)i * (6.907755278982137f / 16.0f));  // 1000^(-i/16)
    const float th = (float)t * f;
    c = __cosf(th);
    s = __sinf(th);
  } else {
    c = 1.0f;
    s = 0.0f;
  }
  float o1 = c * x1 - s * x2;
  float o2 = s * x1 + c * x2;
  if (grp == 0) {
    o1 *= 0.125f;  // 1/sqrt(DH) folded into Q
    o2 *= 0.125f;
    Q[orow + i] = (__bf16)o1;
    Q[orow + 32 + i] = (__bf16)o2;
  } else {
    Kk[orow + i] = (__bf16)o1;
    Kk[orow + 32 + i] = (__bf16)o2;
  }
}

// ---------------------------------------------------------------------------
// Flash attention (causal), bf16 WMMA.  Block = 8 waves = 128 q-rows per
// (b,h); wave = 16 q-rows.  K tiles of 32 keys staged via async-to-LDS
// (stride 72 rows: conflict-free); V staged transposed (stride 40).
// P bounced through per-wave LDS to convert WMMA C-layout -> A-layout
// (same-wave DS ops are in-order; wave_barrier stops compiler reordering).
// The causality guard goes through readfirstlane -> scalar branch, so EXEC
// is untouched (all ones) around every WMMA as the ISA requires.
// ---------------------------------------------------------------------------
__global__ __launch_bounds__(256) void k_attn(const __bf16* __restrict__ Q,
                                              const __bf16* __restrict__ K,
                                              const __bf16* __restrict__ V,
                                              __bf16* __restrict__ ctx) {
  __shared__ __bf16 sK[32 * 72];      // 4.5 KB
  __shared__ __bf16 sV[64 * 40];      // 5 KB   (transposed: [d][key])
  __shared__ __bf16 sP[8 * 16 * 40];  // 10 KB  (per-wave 16x32 P tile)

  const int tid = threadIdx.x;
  const int lane = tid & 31;
  const int wave = tid >> 5;
  const int lr = lane & 15;
  const int hs = lane >> 4;
  const int bh = blockIdx.x;  // b*H + h
  const int b = bh / H_, h = bh % H_;
  const int qblk = blockIdx.y * 128;
  const int qrow0 = qblk + wave * 16;
  const size_t base = (size_t)bh * T_ * DH_;

  // Q fragments (A-layout): two 16x32 frags covering DH=64
  bf16x16 qf[2];
#pragma unroll
  for (int f = 0; f < 2; ++f) {
    const __bf16* p = Q + base + (size_t)(qrow0 + lr) * DH_ + f * 32 + hs * 8;
    qf[f] = cat8(*(const bf16x8*)p, *(const bf16x8*)(p + 16));
  }

  floatx8 acc[4] = {};
  float mrow[8], lrow[8];
#pragma unroll
  for (int v = 0; v < 8; ++v) {
    mrow[v] = -3.0e38f;
    lrow[v] = 0.0f;
  }

  const int srow = tid >> 3, sch = (tid & 7) * 8;  // staging coords
  const int nkt = (qblk + 127) / 32 + 1;           // key tiles (block bound)
  for (int kt = 0; kt < nkt; ++kt) {
    const int kb = kt * 32;
    // stage K tile [32 keys][64 d] -> async memory->LDS, 16B per thread
    async_load_b128(lds_addr(&sK[srow * 72 + sch]),
                    K + base + (size_t)(kb + srow) * DH_ + sch);
    {  // stage V transposed: sV[d][key] (VGPR path: needs transpose)
      bf16x8 vv = *(const bf16x8*)(V + base + (size_t)(kb + srow) * DH_ + sch);
#pragma unroll
      for (int e = 0; e < 8; ++e) sV[(sch + e) * 40 + srow] = vv[e];
    }
    wait_async0();
    __syncthreads();

    if (__builtin_amdgcn_readfirstlane((int)(kb <= qrow0 + 15))) {
      // ---- S = Q K^T : two 16-key subtiles ----
      floatx8 s[2];
#pragma unroll
      for (int sub = 0; sub < 2; ++sub) {
        floatx8 c = {};
#pragma unroll
        for (int f = 0; f < 2; ++f) {
          const __bf16* p = &sK[(sub * 16 + lr) * 72 + f * 32 + hs * 8];
          c = wmma_bf16(qf[f], cat8(*(const bf16x8*)p, *(const bf16x8*)(p + 16)), c);
        }
        s[sub] = c;
      }
      // ---- causal mask + online softmax (row = hs*8+v, col = lr across lanes)
      float alpha[8];
      const int c0 = kb + lr, c1 = kb + 16 + lr;
#pragma unroll
      for (int v = 0; v < 8; ++v) {
        const int qr = qrow0 + hs * 8 + v;
        const float s0 = (c0 <= qr) ? s[0][v] : -3.0e38f;
        const float s1 = (c1 <= qr) ? s[1][v] : -3.0e38f;
        float mx = fmaxf(s0, s1);
#pragma unroll
        for (int o = 1; o < 16; o <<= 1) mx = fmaxf(mx, __shfl_xor(mx, o, 32));
        const float mn = fmaxf(mrow[v], mx);
        const float a = __expf(mrow[v] - mn);
        const float p0 = __expf(s0 - mn);
        const float p1 = __expf(s1 - mn);
        float ps = p0 + p1;
#pragma unroll
        for (int o = 1; o < 16; o <<= 1) ps += __shfl_xor(ps, o, 32);
        lrow[v] = lrow[v] * a + ps;
        mrow[v] = mn;
        alpha[v] = a;
        sP[wave * 640 + (hs * 8 + v) * 40 + lr] = (__bf16)p0;
        sP[wave * 640 + (hs * 8 + v) * 40 + 16 + lr] = (__bf16)p1;
      }
#pragma unroll
      for (int j = 0; j < 4; ++j)
#pragma unroll
        for (int v = 0; v < 8; ++v) acc[j][v] *= alpha[v];
      __builtin_amdgcn_wave_barrier();
      // P fragment (A-layout 16x32) read back from per-wave LDS
      const __bf16* pp = &sP[wave * 640 + lr * 40 + hs * 8];
      const bf16x16 pf = cat8(*(const bf16x8*)pp, *(const bf16x8*)(pp + 16));
#pragma unroll
      for (int j = 0; j < 4; ++j) {  // ctx += P * V over 4 d-subtiles
        const __bf16* p = &sV[(j * 16 + lr) * 40 + hs * 8];
        acc[j] = wmma_bf16(pf, cat8(*(const bf16x8*)p, *(const bf16x8*)(p + 16)),
                           acc[j]);
      }
    }
    __syncthreads();
  }

  // ---- finalize: ctx[b][t][h][d] (== [BT][D]) in bf16 for the Wout GEMM
#pragma unroll
  for (int j = 0; j < 4; ++j)
#pragma unroll
    for (int v = 0; v < 8; ++v) {
      const int qr = qrow0 + hs * 8 + v;
      const float o = acc[j][v] / lrow[v];
      ctx[(size_t)(b * T_ + qr) * D_ + h * DH_ + j * 16 + lr] = (__bf16)o;
    }
}

// ---------------------------------------------------------------------------
// Host orchestration
// ---------------------------------------------------------------------------
extern "C" void kernel_launch(void* const* d_in, const int* in_sizes, int n_in,
                              void* d_out, int out_size, void* d_ws,
                              size_t ws_size, hipStream_t stream) {
  (void)in_sizes; (void)n_in; (void)out_size; (void)ws_size;
  const int* xind = (const int*)d_in[0];
  const float* emb = (const float*)d_in[1];
  const float* Wqkv = (const float*)d_in[2];
  const float* Wout = (const float*)d_in[3];
  const float* W1 = (const float*)d_in[4];
  const float* W2 = (const float*)d_in[5];
  const float* lmh = (const float*)d_in[6];
  float* out = (float*)d_out;

  char* ws = (char*)d_ws;
  size_t off = 0;
  auto alloc = [&](size_t bytes) -> char* {
    char* p = ws + off;
    off = (off + bytes + 255) & ~(size_t)255;
    return p;
  };
  __bf16* wqkvb = (__bf16*)alloc((size_t)L_ * 3 * D_ * D_ * 2);
  __bf16* woutb = (__bf16*)alloc((size_t)L_ * D_ * D_ * 2);
  __bf16* w1b = (__bf16*)alloc((size_t)L_ * FF_ * D_ * 2);
  __bf16* w2b = (__bf16*)alloc((size_t)L_ * D_ * FF_ * 2);
  __bf16* lmhb = (__bf16*)alloc((size_t)V_ * D_ * 2);
  float* x = (float*)alloc((size_t)BT_ * D_ * 4);
  __bf16* nb = (__bf16*)alloc((size_t)BT_ * D_ * 2);
  float* qkvf = (float*)alloc((size_t)BT_ * 3 * D_ * 4);
  __bf16* qb = (__bf16*)alloc((size_t)BT_ * D_ * 2);
  __bf16* kb = (__bf16*)alloc((size_t)BT_ * D_ * 2);
  __bf16* vb = (__bf16*)alloc((size_t)BT_ * D_ * 2);
  __bf16* ctxb = (__bf16*)alloc((size_t)BT_ * D_ * 2);
  __bf16* hb = (__bf16*)alloc((size_t)BT_ * FF_ * 2);

  // Weight conversion (once per call; deterministic)
  k_cast_bf16<<<2048, 256, 0, stream>>>(Wqkv, wqkvb, (size_t)L_ * 3 * D_ * D_ / 4);
  k_cast_bf16<<<2048, 256, 0, stream>>>(Wout, woutb, (size_t)L_ * D_ * D_ / 4);
  k_cast_bf16<<<2048, 256, 0, stream>>>(W1, w1b, (size_t)L_ * FF_ * D_ / 4);
  k_cast_bf16<<<2048, 256, 0, stream>>>(W2, w2b, (size_t)L_ * D_ * FF_ / 4);
  k_cast_bf16<<<2048, 256, 0, stream>>>(lmh, lmhb, (size_t)V_ * D_ / 4);

  k_embed<<<BT_, 256, 0, stream>>>(xind, emb, x);

  const dim3 gQKV(3 * D_ / GBN, BT_ / GBM);
  const dim3 gOUT(D_ / GBN, BT_ / GBM);
  const dim3 gFF1(FF_ / GBN, BT_ / GBM);
  const dim3 gLM(V_ / GBN, BT_ / GBM);
  const dim3 gATT(B_ * H_, T_ / 128);

  for (int l = 0; l < L_; ++l) {
    k_rmsnorm<<<BT_, 256, 0, stream>>>(x, nb);
    k_gemm_bf16<0><<<gQKV, 256, 0, stream>>>(
        nb, wqkvb + (size_t)l * 3 * D_ * D_, BT_, 3 * D_, D_, nullptr, qkvf,
        nullptr);
    k_rope_split<<<B_ * H_ * T_, 64, 0, stream>>>(qkvf, qb, kb, vb);
    k_attn<<<gATT, 256, 0, stream>>>(qb, kb, vb, ctxb);
    k_gemm_bf16<1><<<gOUT, 256, 0, stream>>>(ctxb,
                                             woutb + (size_t)l * D_ * D_, BT_,
                                             D_, D_, x, nullptr, nullptr);
    k_rmsnorm<<<BT_, 256, 0, stream>>>(x, nb);
    k_gemm_bf16<2><<<gFF1, 256, 0, stream>>>(nb, w1b + (size_t)l * FF_ * D_,
                                             BT_, FF_, D_, nullptr, nullptr,
                                             hb);
    k_gemm_bf16<1><<<gOUT, 256, 0, stream>>>(hb, w2b + (size_t)l * D_ * FF_,
                                             BT_, D_, FF_, x, nullptr,
                                             nullptr);
  }
  k_rmsnorm<<<BT_, 256, 0, stream>>>(x, nb);
  k_gemm_bf16<0><<<gLM, 256, 0, stream>>>(nb, lmhb, BT_, V_, D_, nullptr, out,
                                          nullptr);
}
